// ModernTransformerLayer_13305808683522
// MI455X (gfx1250) — matrix operand order repaired and checked by reading, not compiled, not run on hardware
//
#include <hip/hip_runtime.h>

typedef __attribute__((ext_vector_type(16))) __bf16 v16bf;
typedef __attribute__((ext_vector_type(8)))  float  v8f;

#define Bc     4
#define Sc     2048
#define Hc     1024
#define NHc    16
#define HDc    64
#define INNERc 4096
#define Mc     (Bc * Sc)   // 8192 rows

#define LDP 36             // padded LDS row stride (32 data + 4 pad) in bf16 elems

union Frag {
    unsigned int u[8];
    v16bf v;
};

__device__ __forceinline__ unsigned short f2bf(float f) {
    unsigned int u = __builtin_bit_cast(unsigned int, f);
    unsigned int r = u + 0x7FFFu + ((u >> 16) & 1u);   // round-to-nearest-even
    return (unsigned short)(r >> 16);
}

// Async copy 8 bytes global -> LDS (ASYNCcnt-tracked, no VGPR data round-trip)
__device__ __forceinline__ void async_g2l_b64(unsigned lds_off, const void* gaddr) {
    asm volatile("global_load_async_to_lds_b64 %0, %1, off"
                 :: "v"(lds_off), "v"(gaddr) : "memory");
}
__device__ __forceinline__ void wait_async0() {
    asm volatile("s_wait_asynccnt 0x0" ::: "memory");
}

// ---------------------------------------------------------------------------
// fp32 -> bf16 elementwise (weights)
// ---------------------------------------------------------------------------
__global__ void k_f32_to_bf16(const float* __restrict__ in,
                              unsigned short* __restrict__ out, int n) {
    int i = blockIdx.x * blockDim.x + threadIdx.x;
    if (i < n) out[i] = f2bf(in[i]);
}

// ---------------------------------------------------------------------------
// RMSNorm: one row (H=1024) per 256-thread block, output bf16
// ---------------------------------------------------------------------------
__global__ __launch_bounds__(256) void k_rmsnorm_bf16(
    const float* __restrict__ x, const float* __restrict__ w,
    unsigned short* __restrict__ out) {
    __shared__ float red[8];
    int row = blockIdx.x;
    const float* xr = x + (size_t)row * Hc;
    float s = 0.f;
    for (int c = threadIdx.x; c < Hc; c += 256) {
        float v = xr[c];
        s += v * v;
    }
    for (int off = 16; off; off >>= 1) s += __shfl_xor(s, off, 32);
    if ((threadIdx.x & 31) == 0) red[threadIdx.x >> 5] = s;
    __syncthreads();
    float tot = 0.f;
    for (int i = 0; i < 8; i++) tot += red[i];
    float scale = rsqrtf(tot / (float)Hc + 1e-8f);
    for (int c = threadIdx.x; c < Hc; c += 256)
        out[(size_t)row * Hc + c] = f2bf(xr[c] * scale * w[c]);
}

// ---------------------------------------------------------------------------
// Generic bf16 GEMM: C[M,N] = A[M,K] @ B[K,N], f32 accumulate via WMMA.
// Block tile 128x128, 8 waves, wave tile 64x32 (4x2 WMMA accums).
// A tile staged via async global->LDS DMA; B tile transposed through VGPRs.
// MODE 0: write bf16 to outB.  MODE 1: outF = acc + res.  (compile-time)
// ---------------------------------------------------------------------------
template <int MODE>
__global__ __launch_bounds__(256) void k_gemm_bf16(
    const unsigned short* __restrict__ A, const unsigned short* __restrict__ Bm,
    int Kdim, int Ndim,
    const float* __restrict__ res, float* __restrict__ outF,
    unsigned short* __restrict__ outB) {
    __shared__ __align__(16) unsigned short Ah[128 * LDP];
    __shared__ __align__(16) unsigned short Bt[128 * LDP];

    int t = threadIdx.x;
    int lane = t & 31, wave = t >> 5;
    int wm = wave >> 2, wn = wave & 3;          // 2 x 4 wave grid
    int half = lane >> 4, l16 = lane & 15;
    int m0 = blockIdx.y * 128, n0 = blockIdx.x * 128;
    unsigned ahBase = (unsigned)(size_t)Ah;

    v8f acc[4][2];
    for (int mt = 0; mt < 4; mt++)
        for (int nt = 0; nt < 2; nt++)
            for (int i = 0; i < 8; i++) acc[mt][nt][i] = 0.f;

    for (int k0 = 0; k0 < Kdim; k0 += 32) {
        __syncthreads();
        // A tile 128x32 bf16: async DMA, 1024 chunks of 8B (4 per thread)
        const char* Agb = (const char*)(A + (size_t)m0 * Kdim + k0);
        for (int i = 0; i < 4; i++) {
            int l = i * 256 + t;
            int r = l >> 3, c8 = l & 7;                 // row, 8B-chunk in row
            async_g2l_b64(ahBase + (unsigned)(r * LDP + c8 * 4) * 2u,
                          Agb + (size_t)r * Kdim * 2 + c8 * 8);
        }
        // B tile 32x128, stored transposed BT[n][k]; dword global loads
        for (int i = 0; i < 8; i++) {
            int l = i * 256 + t;
            int nn2 = l & 63, kk = l >> 6;              // dword col, k row
            unsigned v = *(const unsigned int*)&Bm[(size_t)(k0 + kk) * Ndim +
                                                   n0 + nn2 * 2];
            Bt[(nn2 * 2) * LDP + kk]     = (unsigned short)v;
            Bt[(nn2 * 2 + 1) * LDP + kk] = (unsigned short)(v >> 16);
        }
        wait_async0();
        __syncthreads();

        Frag a[4], b[2];
        for (int mt = 0; mt < 4; mt++) {
            int r = wm * 64 + mt * 16 + l16;
            for (int j = 0; j < 8; j++) {
                int kk = (j < 4 ? 2 * j : 8 + 2 * j) + (half ? 8 : 0);
                a[mt].u[j] = *(const unsigned int*)&Ah[r * LDP + kk];
            }
        }
        for (int nt = 0; nt < 2; nt++) {
            int n = wn * 32 + nt * 16 + l16;
            for (int j = 0; j < 8; j++) {
                int kk = (j < 4 ? 2 * j : 8 + 2 * j) + (half ? 8 : 0);
                b[nt].u[j] = *(const unsigned int*)&Bt[n * LDP + kk];
            }
        }
        for (int mt = 0; mt < 4; mt++)
            for (int nt = 0; nt < 2; nt++)
                acc[mt][nt] = __builtin_amdgcn_wmma_f32_16x16x32_bf16(
                    false, a[mt].v, false, b[nt].v, (short)0, acc[mt][nt],
                    false, false);
    }

    for (int mt = 0; mt < 4; mt++)
        for (int nt = 0; nt < 2; nt++) {
            int col = n0 + wn * 32 + nt * 16 + l16;
            int row0 = m0 + wm * 64 + mt * 16 + (half ? 8 : 0);
            size_t base = (size_t)row0 * Ndim + col;
            for (int i = 0; i < 8; i++) {
                size_t idx = base + (size_t)i * Ndim;
                float v = acc[mt][nt][i];
                if (MODE == 0)
                    outB[idx] = f2bf(v);
                else
                    outF[idx] = v + res[idx];
            }
        }
}

// ---------------------------------------------------------------------------
// Fused SwiGLU GEMM: g = silu(A@W1) * (A@W2), bf16 output.
// Block tile 128x64, 8 waves, wave tile 64x16, dual accumulators.
// ---------------------------------------------------------------------------
__global__ __launch_bounds__(256) void k_gemm_swiglu(
    const unsigned short* __restrict__ A,
    const unsigned short* __restrict__ B1, const unsigned short* __restrict__ B2,
    int Kdim, int Ndim, unsigned short* __restrict__ outG) {
    __shared__ __align__(16) unsigned short Ah[128 * LDP];
    __shared__ __align__(16) unsigned short B1t[64 * LDP];
    __shared__ __align__(16) unsigned short B2t[64 * LDP];

    int t = threadIdx.x;
    int lane = t & 31, wave = t >> 5;
    int wm = wave >> 2, wn = wave & 3;
    int half = lane >> 4, l16 = lane & 15;
    int m0 = blockIdx.y * 128, n0 = blockIdx.x * 64;
    unsigned ahBase = (unsigned)(size_t)Ah;

    v8f acc1[4], acc2[4];
    for (int mt = 0; mt < 4; mt++)
        for (int i = 0; i < 8; i++) { acc1[mt][i] = 0.f; acc2[mt][i] = 0.f; }

    for (int k0 = 0; k0 < Kdim; k0 += 32) {
        __syncthreads();
        const char* Agb = (const char*)(A + (size_t)m0 * Kdim + k0);
        for (int i = 0; i < 4; i++) {
            int l = i * 256 + t;
            int r = l >> 3, c8 = l & 7;
            async_g2l_b64(ahBase + (unsigned)(r * LDP + c8 * 4) * 2u,
                          Agb + (size_t)r * Kdim * 2 + c8 * 8);
        }
        for (int i = 0; i < 4; i++) {
            int l = i * 256 + t;
            int nn2 = l & 31, kk = l >> 5;
            size_t gi = (size_t)(k0 + kk) * Ndim + n0 + nn2 * 2;
            unsigned v1 = *(const unsigned int*)&B1[gi];
            unsigned v2 = *(const unsigned int*)&B2[gi];
            B1t[(nn2 * 2) * LDP + kk]     = (unsigned short)v1;
            B1t[(nn2 * 2 + 1) * LDP + kk] = (unsigned short)(v1 >> 16);
            B2t[(nn2 * 2) * LDP + kk]     = (unsigned short)v2;
            B2t[(nn2 * 2 + 1) * LDP + kk] = (unsigned short)(v2 >> 16);
        }
        wait_async0();
        __syncthreads();

        Frag a[4], b1, b2;
        for (int mt = 0; mt < 4; mt++) {
            int r = wm * 64 + mt * 16 + l16;
            for (int j = 0; j < 8; j++) {
                int kk = (j < 4 ? 2 * j : 8 + 2 * j) + (half ? 8 : 0);
                a[mt].u[j] = *(const unsigned int*)&Ah[r * LDP + kk];
            }
        }
        {
            int n = wn * 16 + l16;
            for (int j = 0; j < 8; j++) {
                int kk = (j < 4 ? 2 * j : 8 + 2 * j) + (half ? 8 : 0);
                b1.u[j] = *(const unsigned int*)&B1t[n * LDP + kk];
                b2.u[j] = *(const unsigned int*)&B2t[n * LDP + kk];
            }
        }
        for (int mt = 0; mt < 4; mt++) {
            acc1[mt] = __builtin_amdgcn_wmma_f32_16x16x32_bf16(
                false, a[mt].v, false, b1.v, (short)0, acc1[mt], false, false);
            acc2[mt] = __builtin_amdgcn_wmma_f32_16x16x32_bf16(
                false, a[mt].v, false, b2.v, (short)0, acc2[mt], false, false);
        }
    }

    for (int mt = 0; mt < 4; mt++) {
        int col = n0 + wn * 16 + l16;
        int row0 = m0 + wm * 64 + mt * 16 + (half ? 8 : 0);
        size_t base = (size_t)row0 * Ndim + col;
        for (int i = 0; i < 8; i++) {
            float v1 = acc1[mt][i];
            float v2 = acc2[mt][i];
            float g = (v1 / (1.f + __expf(-v1))) * v2;   // silu(v1) * v2
            outG[base + (size_t)i * Ndim] = f2bf(g);
        }
    }
}

// ---------------------------------------------------------------------------
// Flash attention: one wave32 per (b, head, 32 q-rows). K staged via async
// DMA, V transposed through VGPRs; Q@K^T and P@V via WMMA; online softmax.
// ---------------------------------------------------------------------------
__global__ __launch_bounds__(32) void k_attn(
    const unsigned short* __restrict__ qg, const unsigned short* __restrict__ kg,
    const unsigned short* __restrict__ vg, const float* __restrict__ mask,
    unsigned short* __restrict__ ctx) {
    __shared__ __align__(16) unsigned short Kt[32 * 68];   // [pos][d], pad 68
    __shared__ __align__(16) unsigned short Vt[64 * LDP];  // transposed [d][pos]
    __shared__ __align__(16) unsigned short Pt[32 * LDP];  // probs [row][pos]

    int lane = threadIdx.x;
    int half = lane >> 4, l16 = lane & 15;
    int qb = blockIdx.x * 32;
    int bh = blockIdx.y;
    int b = bh >> 4, hh = bh & 15;
    size_t headoff = (size_t)hh * HDc;
    unsigned ktBase = (unsigned)(size_t)&Kt[lane * 68];

    // Preload Q fragments (2 row tiles x 2 K-steps of 32 over HD=64)
    Frag aq[2][2];
    for (int mt = 0; mt < 2; mt++)
        for (int ks = 0; ks < 2; ks++) {
            int row = qb + mt * 16 + l16;
            const unsigned short* qrow =
                qg + ((size_t)(b * Sc + row)) * Hc + headoff;
            for (int j = 0; j < 8; j++) {
                int dd = ks * 32 + (j < 4 ? 2 * j : 8 + 2 * j) + (half ? 8 : 0);
                aq[mt][ks].u[j] = *(const unsigned int*)&qrow[dd];
            }
        }

    float mrun[2][8], lrun[2][8];
    v8f cacc[2][4];
    for (int mt = 0; mt < 2; mt++)
        for (int i = 0; i < 8; i++) { mrun[mt][i] = -3.0e38f; lrun[mt][i] = 0.f; }
    for (int mt = 0; mt < 2; mt++)
        for (int nt = 0; nt < 4; nt++)
            for (int i = 0; i < 8; i++) cacc[mt][nt][i] = 0.f;

    for (int kb = 0; kb < Sc; kb += 32) {
        // Stage K block [32 pos][64 d] via async DMA (each lane = one row)
        {
            const char* krowb = (const char*)(
                kg + ((size_t)(b * Sc + kb + lane)) * Hc + headoff);
            for (int i = 0; i < 16; i++)
                async_g2l_b64(ktBase + (unsigned)(i * 8), krowb + i * 8);
            // V block transposed [64 d][32 pos] via VGPR path
            const unsigned short* vrow =
                vg + ((size_t)(b * Sc + kb + lane)) * Hc + headoff;
            for (int d = 0; d < 64; d++) Vt[d * LDP + lane] = vrow[d];
        }
        wait_async0();
        __syncthreads();

        // Scores: S = Q @ K^T  (2x2 accumulators, 2 K-steps)
        v8f sacc[2][2];
        for (int mt = 0; mt < 2; mt++)
            for (int nt = 0; nt < 2; nt++)
                for (int i = 0; i < 8; i++) sacc[mt][nt][i] = 0.f;
        for (int ks = 0; ks < 2; ks++) {
            Frag bk[2];
            for (int nt = 0; nt < 2; nt++) {
                int n = nt * 16 + l16;
                for (int j = 0; j < 8; j++) {
                    int dd = ks * 32 + (j < 4 ? 2 * j : 8 + 2 * j) + (half ? 8 : 0);
                    bk[nt].u[j] = *(const unsigned int*)&Kt[n * 68 + dd];
                }
            }
            for (int mt = 0; mt < 2; mt++)
                for (int nt = 0; nt < 2; nt++)
                    sacc[mt][nt] = __builtin_amdgcn_wmma_f32_16x16x32_bf16(
                        false, aq[mt][ks].v, false, bk[nt].v, (short)0,
                        sacc[mt][nt], false, false);
        }

        // Online softmax update + write P to LDS
        for (int mt = 0; mt < 2; mt++) {
            for (int i = 0; i < 8; i++) {
                int qrow = qb + mt * 16 + i + (half ? 8 : 0);
                float s0 = sacc[mt][0][i] * 0.125f +
                           mask[(size_t)qrow * Sc + kb + l16];
                float s1 = sacc[mt][1][i] * 0.125f +
                           mask[(size_t)qrow * Sc + kb + 16 + l16];
                float rmax = fmaxf(s0, s1);
                for (int off = 8; off; off >>= 1)
                    rmax = fmaxf(rmax, __shfl_xor(rmax, off, 32));
                float mold = mrun[mt][i];
                float mnew = fmaxf(mold, rmax);
                float alpha = __expf(mold - mnew);
                float p0 = __expf(s0 - mnew);
                float p1 = __expf(s1 - mnew);
                float psum = p0 + p1;
                for (int off = 8; off; off >>= 1)
                    psum += __shfl_xor(psum, off, 32);
                lrun[mt][i] = lrun[mt][i] * alpha + psum;
                mrun[mt][i] = mnew;
                for (int nt = 0; nt < 4; nt++)
                    cacc[mt][nt][i] = cacc[mt][nt][i] * alpha;
                int prow = mt * 16 + i + (half ? 8 : 0);
                Pt[prow * LDP + l16] = f2bf(p0);
                Pt[prow * LDP + 16 + l16] = f2bf(p1);
            }
        }
        __syncthreads();

        // ctx += P(32x32) @ V(32x64)
        Frag ap[2];
        for (int mt = 0; mt < 2; mt++) {
            int r = mt * 16 + l16;
            for (int j = 0; j < 8; j++) {
                int kk = (j < 4 ? 2 * j : 8 + 2 * j) + (half ? 8 : 0);
                ap[mt].u[j] = *(const unsigned int*)&Pt[r * LDP + kk];
            }
        }
        for (int nt = 0; nt < 4; nt++) {
            Frag bv;
            int d = nt * 16 + l16;
            for (int j = 0; j < 8; j++) {
                int kk = (j < 4 ? 2 * j : 8 + 2 * j) + (half ? 8 : 0);
                bv.u[j] = *(const unsigned int*)&Vt[d * LDP + kk];
            }
            for (int mt = 0; mt < 2; mt++)
                cacc[mt][nt] = __builtin_amdgcn_wmma_f32_16x16x32_bf16(
                    false, ap[mt].v, false, bv.v, (short)0, cacc[mt][nt],
                    false, false);
        }
        __syncthreads();
    }

    // Normalize and write ctx (bf16, [B,S,H] with head offset)
    for (int mt = 0; mt < 2; mt++)
        for (int nt = 0; nt < 4; nt++)
            for (int i = 0; i < 8; i++) {
                int row = qb + mt * 16 + i + (half ? 8 : 0);
                int d = nt * 16 + l16;
                float v = cacc[mt][nt][i] / lrun[mt][i];
                ctx[((size_t)(b * Sc + row)) * Hc + headoff + d] = f2bf(v);
            }
}

// ---------------------------------------------------------------------------
extern "C" void kernel_launch(void* const* d_in, const int* in_sizes, int n_in,
                              void* d_out, int out_size, void* d_ws,
                              size_t ws_size, hipStream_t stream) {
    const float* x    = (const float*)d_in[0];
    const float* mask = (const float*)d_in[1];
    const float* ln1  = (const float*)d_in[2];
    const float* ln2  = (const float*)d_in[3];
    const float* Wq   = (const float*)d_in[4];
    const float* Wk   = (const float*)d_in[5];
    const float* Wv   = (const float*)d_in[6];
    const float* Wo   = (const float*)d_in[7];
    const float* W1   = (const float*)d_in[8];
    const float* W2   = (const float*)d_in[9];
    const float* W3   = (const float*)d_in[10];
    float* out = (float*)d_out;

    char* ws = (char*)d_ws;
    size_t o = 0;
    auto alloc = [&](size_t bytes) {
        void* p = ws + o;
        o += (bytes + 255) & ~(size_t)255;
        return p;
    };
    unsigned short* wq_b  = (unsigned short*)alloc((size_t)Hc * Hc * 2);
    unsigned short* wk_b  = (unsigned short*)alloc((size_t)Hc * Hc * 2);
    unsigned short* wv_b  = (unsigned short*)alloc((size_t)Hc * Hc * 2);
    unsigned short* wo_b  = (unsigned short*)alloc((size_t)Hc * Hc * 2);
    unsigned short* w1_b  = (unsigned short*)alloc((size_t)Hc * INNERc * 2);
    unsigned short* w2_b  = (unsigned short*)alloc((size_t)Hc * INNERc * 2);
    unsigned short* w3_b  = (unsigned short*)alloc((size_t)INNERc * Hc * 2);
    unsigned short* h_b   = (unsigned short*)alloc((size_t)Mc * Hc * 2);
    unsigned short* q_b   = (unsigned short*)alloc((size_t)Mc * Hc * 2);
    unsigned short* k_b   = (unsigned short*)alloc((size_t)Mc * Hc * 2);
    unsigned short* v_b   = (unsigned short*)alloc((size_t)Mc * Hc * 2);
    unsigned short* ctx_b = (unsigned short*)alloc((size_t)Mc * Hc * 2);
    float*          xat   = (float*)alloc((size_t)Mc * Hc * 4);
    unsigned short* g_b   = (unsigned short*)alloc((size_t)Mc * INNERc * 2);

    // 1) weights -> bf16
    int nHH = Hc * Hc, nHI = Hc * INNERc;
    k_f32_to_bf16<<<(nHH + 255) / 256, 256, 0, stream>>>(Wq, wq_b, nHH);
    k_f32_to_bf16<<<(nHH + 255) / 256, 256, 0, stream>>>(Wk, wk_b, nHH);
    k_f32_to_bf16<<<(nHH + 255) / 256, 256, 0, stream>>>(Wv, wv_b, nHH);
    k_f32_to_bf16<<<(nHH + 255) / 256, 256, 0, stream>>>(Wo, wo_b, nHH);
    k_f32_to_bf16<<<(nHI + 255) / 256, 256, 0, stream>>>(W1, w1_b, nHI);
    k_f32_to_bf16<<<(nHI + 255) / 256, 256, 0, stream>>>(W2, w2_b, nHI);
    k_f32_to_bf16<<<(nHI + 255) / 256, 256, 0, stream>>>(W3, w3_b, nHI);

    // 2) h = rmsnorm(x, ln1)  (bf16)
    k_rmsnorm_bf16<<<Mc, 256, 0, stream>>>(x, ln1, h_b);

    // 3) Q/K/V projections (bf16 out)
    dim3 gP(Hc / 128, Mc / 128);
    k_gemm_bf16<0><<<gP, 256, 0, stream>>>(h_b, wq_b, Hc, Hc, nullptr, nullptr, q_b);
    k_gemm_bf16<0><<<gP, 256, 0, stream>>>(h_b, wk_b, Hc, Hc, nullptr, nullptr, k_b);
    k_gemm_bf16<0><<<gP, 256, 0, stream>>>(h_b, wv_b, Hc, Hc, nullptr, nullptr, v_b);

    // 4) attention -> ctx (bf16)
    k_attn<<<dim3(Sc / 32, Bc * NHc), 32, 0, stream>>>(q_b, k_b, v_b, mask, ctx_b);

    // 5) x_attn = x + ctx @ Wo   (f32)
    k_gemm_bf16<1><<<gP, 256, 0, stream>>>(ctx_b, wo_b, Hc, Hc, x, xat, nullptr);

    // 6) h2 = rmsnorm(x_attn, ln2) (bf16, reuse h_b)
    k_rmsnorm_bf16<<<Mc, 256, 0, stream>>>(xat, ln2, h_b);

    // 7) g = silu(h2@W1) * (h2@W2)  (bf16)
    k_gemm_swiglu<<<dim3(INNERc / 64, Mc / 128), 256, 0, stream>>>(
        h_b, w1_b, w2_b, Hc, INNERc, g_b);

    // 8) out = x_attn + g @ W3  (f32)
    k_gemm_bf16<1><<<dim3(Hc / 128, Mc / 128), 256, 0, stream>>>(
        g_b, w3_b, INNERc, Hc, xat, out, nullptr);

    (void)in_sizes; (void)n_in; (void)out_size; (void)ws_size;
}
